// RefNetGCN_84172769068177
// MI455X (gfx1250) — matrix operand order repaired and checked by reading, not compiled, not run on hardware
//
#include <hip/hip_runtime.h>
#include <stdint.h>
#include <math.h>

typedef _Float16 h16;
typedef __attribute__((ext_vector_type(16))) _Float16 v16h;
typedef __attribute__((ext_vector_type(8)))  float    v8f;

#define DEV static __device__ __forceinline__

// ---------------------------------------------------------------------------
// WMMA helpers (CDNA5 v_wmma_f32_16x16x32_f16, wave32)
// A layout (16x32 f16): lane row = lane&15; half i -> K = (lane>>4)*16 + i.
// B layout (32x16 f16): lane col = lane&15; half i -> K = (lane>>4)*16 + i.
// C/D layout (16x16 f32): lane col = lane&15; acc[i] -> row (lane>>4)*8 + i.
// ---------------------------------------------------------------------------
template <typename T>
DEV v16h packA(const T* src, int ld, int row0, int k0) {
  const int ln = threadIdx.x & 31;
  const T* p = src + (size_t)(row0 + (ln & 15)) * ld + k0 + ((ln >> 4) * 16);
  v16h a;
#pragma unroll
  for (int i = 0; i < 8; ++i) a[i] = (_Float16)p[i];
#pragma unroll
  for (int i = 0; i < 8; ++i) a[8 + i] = (_Float16)p[8 + i];
  return a;
}

// B operand from a K-major (K x N, row stride ld) matrix: strided gather.
template <typename T>
DEV v16h packB_km(const T* src, int ld, int k0, int col0) {
  const int ln = threadIdx.x & 31;
  const T* p = src + (size_t)(k0 + ((ln >> 4) * 16)) * ld + col0 + (ln & 15);
  v16h b;
#pragma unroll
  for (int i = 0; i < 16; ++i) b[i] = (_Float16)p[(size_t)i * ld];
  return b;
}

// B operand from an N-major (N x K, row stride ld) matrix: contiguous chunk.
template <typename T>
DEV v16h packB_nk(const T* src, int ld, int k0, int col0) {
  const int ln = threadIdx.x & 31;
  const T* p = src + (size_t)(col0 + (ln & 15)) * ld + k0 + ((ln >> 4) * 16);
  v16h b;
#pragma unroll
  for (int i = 0; i < 16; ++i) b[i] = (_Float16)p[i];
  return b;
}

DEV v8f wmma16(v16h a, v16h b, v8f c) {
  return __builtin_amdgcn_wmma_f32_16x16x32_f16(false, a, false, b,
                                                (short)0, c, false, false);
}

#define BQ 8
#define LQ 80
#define NQ 8192
#define ROWS ((size_t)BQ * NQ)     // 65536 flattened object rows
__constant__ const float kScale = 0.04419417382415922f;  // 1/sqrt(512)

// ---------------------------------------------------------------------------
// K0: zero a u32 region (pT1|pT2|pT3|words_pad are laid out contiguously)
// ---------------------------------------------------------------------------
__global__ void zero_kernel(uint32_t* p, size_t n) {
  size_t i = (size_t)blockIdx.x * blockDim.x + threadIdx.x;
  size_t st = (size_t)gridDim.x * blockDim.x;
  for (; i < n; i += st) p[i] = 0u;
}

// ---------------------------------------------------------------------------
// K1: per-token language pipeline. One block per (b,l). Computes
//  cw-derived weights, etw*cw2 contributions, words (+f16 padded copy),
//  Q, and the folded attention projection Qp = Q @ wk^T, qb = Q . wk_b.
// ---------------------------------------------------------------------------
__global__ __launch_bounds__(256) void lang_kernel(
    const float* lang, const float* mask,
    const float* wj1, const float* wjb1, const float* wj2, const float* wjb2,
    const float* eg1, const float* egb1, const float* eg2, const float* egb2,
    const float* lm1, const float* lmb1, const float* lm2, const float* lmb2,
    const float* wq, const float* wqb, const float* wk, const float* wkb,
    float* Qp, float* qb, float* wnwm, float* fwm, float* etwf, h16* wpad) {
  const int bl = blockIdx.x, b = bl / LQ, l = bl - b * LQ;
  const int t = threadIdx.x;
  __shared__ float sL[768], sH[512], sW[256], sQ[512];
  __shared__ float s64[64], sSm[9], sT4[4], sFw[1];

  for (int i = t; i < 768; i += 256) sL[i] = lang[(size_t)bl * 768 + i];
  __syncthreads();
  const float mk = mask[bl];

  // --- cw = softmax(sigmoid(x@wj1+b)@wj2+b) * mask ---
  if (t < 64) {
    float a = wjb1[t];
    for (int k = 0; k < 768; ++k) a += sL[k] * wj1[k * 64 + t];
    s64[t] = 1.f / (1.f + expf(-a));
  }
  __syncthreads();
  if (t < 4) {
    float a = wjb2[t];
    for (int k = 0; k < 64; ++k) a += s64[k] * wj2[k * 4 + t];
    sT4[t] = a;
  }
  __syncthreads();
  if (t == 0) {
    float m = fmaxf(fmaxf(sT4[0], sT4[1]), fmaxf(sT4[2], sT4[3]));
    float e0 = expf(sT4[0] - m), e1 = expf(sT4[1] - m);
    float e2 = expf(sT4[2] - m), e3 = expf(sT4[3] - m);
    float inv = 1.f / (e0 + e1 + e2 + e3);
    float c0 = e0 * inv * mk, c1 = e1 * inv * mk, c2 = e2 * inv * mk;
    wnwm[bl] = c0 + c1;
    fwm[bl] = c2;
    sFw[0] = c2;
  }
  __syncthreads();

  // --- etw = softmax(mlp_eg(x)); contribution etw*cw2 ---
  if (t < 64) {
    float a = egb1[t];
    for (int k = 0; k < 768; ++k) a += sL[k] * eg1[k * 64 + t];
    s64[t] = a > 0.f ? a : 0.01f * a;
  }
  __syncthreads();
  if (t < 9) {
    float a = egb2[t];
    for (int k = 0; k < 64; ++k) a += s64[k] * eg2[k * 9 + t];
    sSm[t] = a;
  }
  __syncthreads();
  if (t == 0) {
    float m = -INFINITY;
    for (int j = 0; j < 9; ++j) m = fmaxf(m, sSm[j]);
    float s = 0.f;
    for (int j = 0; j < 9; ++j) { sSm[j] = expf(sSm[j] - m); s += sSm[j]; }
    float inv = 1.f / s;
    for (int j = 0; j < 9; ++j) sSm[j] *= inv;
  }
  __syncthreads();
  if (t < 9) etwf[bl * 9 + t] = sSm[t] * sFw[0];

  // --- words = lrelu(x@lm1+b)@lm2+b, masked ---
  for (int u = t; u < 512; u += 256) {
    float a = lmb1[u];
    for (int k = 0; k < 768; ++k) a += sL[k] * lm1[k * 512 + u];
    sH[u] = a > 0.f ? a : 0.01f * a;
  }
  __syncthreads();
  {
    float a = lmb2[t];
    for (int k = 0; k < 512; ++k) a += sH[k] * lm2[k * 256 + t];
    a *= mk;
    sW[t] = a;
    wpad[((size_t)b * 96 + l) * 256 + t] = (h16)a;
  }
  __syncthreads();

  // --- Q = words @ wq + b ---
  for (int d = t; d < 512; d += 256) {
    float a = wqb[d];
    for (int k = 0; k < 256; ++k) a += sW[k] * wq[k * 512 + d];
    sQ[d] = a;
  }
  __syncthreads();

  // --- Qp = Q @ wk^T ; qb = Q . wk_b (folds away the K projection) ---
  {
    const float* wr = wk + (size_t)t * 512;
    float a = 0.f;
    for (int d = 0; d < 512; ++d) a += sQ[d] * wr[d];
    Qp[(size_t)bl * 256 + t] = a;
  }
  if (t == 0) {
    float a = 0.f;
    for (int d = 0; d < 512; ++d) a += sQ[d] * wkb[d];
    qb[bl] = a;
  }
}

// ---------------------------------------------------------------------------
// K3: object MLPs v = lrelu(obj@w1+b1)@w2+b2, f16 output. One wave per
// 16-row tile; layer-1 activations repacked through LDS (intra-wave).
// blockIdx.y selects which of f1/f2/f3.
// ---------------------------------------------------------------------------
__global__ __launch_bounds__(256) void vmlp_kernel(
    const float* obj,
    const float* w1a, const float* b1a, const float* w2a, const float* b2a,
    const float* w1b, const float* b1b, const float* w2b, const float* b2b,
    const float* w1c, const float* b1c, const float* w2c, const float* b2c,
    h16* v1, h16* v2, h16* v3) {
  const int which = blockIdx.y;
  const float* w1 = which == 0 ? w1a : which == 1 ? w1b : w1c;
  const float* b1 = which == 0 ? b1a : which == 1 ? b1b : b1c;
  const float* w2 = which == 0 ? w2a : which == 1 ? w2b : w2c;
  const float* b2 = which == 0 ? b2a : which == 1 ? b2b : b2c;
  h16* vout = which == 0 ? v1 : which == 1 ? v2 : v3;

  __shared__ h16 act[8][16 * 128];
  const int wv = threadIdx.x >> 5;
  const int ln = threadIdx.x & 31;
  const int tile = blockIdx.x * 8 + wv;   // 0..4095
  const int r0 = tile * 16;
  const int colLn = ln & 15, rB = (ln >> 4) * 8;

  v16h a = packA(obj, 32, r0, 0);         // 16x32 rows of obj_feat
#pragma unroll
  for (int j = 0; j < 8; ++j) {           // layer 1: 128 cols
    v16h bm = packB_km(w1, 128, 0, j * 16);
    v8f c = {};
    c = wmma16(a, bm, c);
    const float bias = b1[j * 16 + colLn];
#pragma unroll
    for (int i = 0; i < 8; ++i) {
      float x = c[i] + bias;
      x = x > 0.f ? x : 0.01f * x;        // leaky relu
      act[wv][(rB + i) * 128 + j * 16 + colLn] = (h16)x;
    }
  }
  for (int ct = 0; ct < 16; ++ct) {       // layer 2: 256 cols, K=128
    v8f c = {};
#pragma unroll
    for (int kt = 0; kt < 4; ++kt) {
      v16h a2 = packA(&act[wv][0], 128, 0, kt * 32);
      v16h bm = packB_km(w2, 256, kt * 32, ct * 16);
      c = wmma16(a2, bm, c);
    }
    const float bias = b2[ct * 16 + colLn];
#pragma unroll
    for (int i = 0; i < 8; ++i)
      vout[(size_t)(r0 + rB + i) * 256 + ct * 16 + colLn] = (h16)(c[i] + bias);
  }
}

// ---------------------------------------------------------------------------
// K4: scores[b,l,n] = (nodes[n,:] . Qp[b,l,:] + qb[b,l]) * SCALE
// M = 65536 rows (flattened), N = 80 (5 tiles), K = 256 (8 wmma chunks).
// ---------------------------------------------------------------------------
template <typename T>
__global__ __launch_bounds__(256) void gemm1_kernel(
    const T* nodes, const float* Qp, const float* qb, float* scores) {
  const int wv = threadIdx.x >> 5, ln = threadIdx.x & 31;
  const int tile = blockIdx.x * 8 + wv;   // 0..4095
  const int r0 = tile * 16;
  const int b = r0 >> 13;
  const int n0 = r0 & (NQ - 1);
  const float* QpB = Qp + (size_t)b * LQ * 256;

  v8f acc[5];
#pragma unroll
  for (int lt = 0; lt < 5; ++lt) acc[lt] = (v8f){};
#pragma unroll
  for (int kt = 0; kt < 8; ++kt) {
    v16h a = packA(nodes, 256, r0, kt * 32);
#pragma unroll
    for (int lt = 0; lt < 5; ++lt) {
      v16h bm = packB_nk(QpB, 256, kt * 32, lt * 16);  // B[k][l] = Qp[l][k]
      acc[lt] = wmma16(a, bm, acc[lt]);
    }
  }
  const int colLn = ln & 15, rB = (ln >> 4) * 8;
#pragma unroll
  for (int lt = 0; lt < 5; ++lt) {
    const int l = lt * 16 + colLn;
    const float q = qb[b * LQ + l];
    float* srow = scores + ((size_t)(b * LQ + l)) * NQ + n0 + rB;
#pragma unroll
    for (int i = 0; i < 8; ++i) srow[i] = (acc[lt][i] + q) * kScale;
  }
}

// ---------------------------------------------------------------------------
// K5: double softmax over n (8192) per (b,l) row, scale by mask*rowW,
// write transposed f16 probabilities pT[(b*8192+n)*96 + l].
// max(softmax1) == 1/sum1 exactly, so pass 2 needs only one reduction.
// ---------------------------------------------------------------------------
__global__ __launch_bounds__(256) void softmax2_kernel(
    const float* scores, const float* rowW, const float* mask, h16* pT) {
  const int bl = blockIdx.x, b = bl / LQ, l = bl - b * LQ;
  const float* s = scores + (size_t)bl * NQ;
  const int t = threadIdx.x;
  __shared__ float red[256];

  float v[32];
#pragma unroll
  for (int i = 0; i < 32; ++i) v[i] = s[i * 256 + t];

  float m = -INFINITY;
#pragma unroll
  for (int i = 0; i < 32; ++i) m = fmaxf(m, v[i]);
  red[t] = m; __syncthreads();
  for (int o = 128; o > 0; o >>= 1) {
    if (t < o) red[t] = fmaxf(red[t], red[t + o]);
    __syncthreads();
  }
  m = red[0]; __syncthreads();

  float se = 0.f;
#pragma unroll
  for (int i = 0; i < 32; ++i) { v[i] = expf(v[i] - m); se += v[i]; }
  red[t] = se; __syncthreads();
  for (int o = 128; o > 0; o >>= 1) {
    if (t < o) red[t] += red[t + o];
    __syncthreads();
  }
  const float inv1 = 1.f / red[0]; __syncthreads();

  float s2 = 0.f;
#pragma unroll
  for (int i = 0; i < 32; ++i) { v[i] = expf(v[i] * inv1 - inv1); s2 += v[i]; }
  red[t] = s2; __syncthreads();
  for (int o = 128; o > 0; o >>= 1) {
    if (t < o) red[t] += red[t + o];
    __syncthreads();
  }
  const float f = rowW[bl] * mask[bl] / red[0];

#pragma unroll
  for (int i = 0; i < 32; ++i) {
    const size_t n = (size_t)i * 256 + t;
    pT[((size_t)b * NQ + n) * 96 + l] = (h16)(v[i] * f);
  }
}

// ---------------------------------------------------------------------------
// K6: fusion[n,c] = sum_i (pT_i^T @ words)[n,c] * v_i[n,c]
// M = 65536, K = 96 (80 + zero pad, 3 wmma chunks), N = 256 (16 tiles).
// ---------------------------------------------------------------------------
__global__ __launch_bounds__(256) void fuse_kernel(
    const h16* pT1, const h16* pT2, const h16* pT3, const h16* wpad,
    const h16* v1, const h16* v2, const h16* v3, float* fusion) {
  const int wv = threadIdx.x >> 5, ln = threadIdx.x & 31;
  const int tile = blockIdx.x * 8 + wv;
  const int r0 = tile * 16;
  const int b = r0 >> 13;
  const h16* wpb = wpad + (size_t)b * 96 * 256;
  const int colLn = ln & 15, rB = (ln >> 4) * 8;

  for (int ct = 0; ct < 16; ++ct) {
    v8f c1 = {}, c2 = {}, c3 = {};
#pragma unroll
    for (int kt = 0; kt < 3; ++kt) {
      v16h bm = packB_km(wpb, 256, kt * 32, ct * 16);
      c1 = wmma16(packA(pT1, 96, r0, kt * 32), bm, c1);
      c2 = wmma16(packA(pT2, 96, r0, kt * 32), bm, c2);
      c3 = wmma16(packA(pT3, 96, r0, kt * 32), bm, c3);
    }
#pragma unroll
    for (int i = 0; i < 8; ++i) {
      const size_t off = (size_t)(r0 + rB + i) * 256 + ct * 16 + colLn;
      fusion[off] = c1[i] * (float)v1[off] + c2[i] * (float)v2[off] +
                    c3[i] * (float)v3[off];
    }
  }
}

// ---------------------------------------------------------------------------
// K7: fusion_edge[n,c] = (pT4^T @ words)[n,c] * fusion[n,c]
// ---------------------------------------------------------------------------
__global__ __launch_bounds__(256) void edgefuse_kernel(
    const h16* pT, const h16* wpad, const float* fusion, float* fe) {
  const int wv = threadIdx.x >> 5, ln = threadIdx.x & 31;
  const int tile = blockIdx.x * 8 + wv;
  const int r0 = tile * 16;
  const int b = r0 >> 13;
  const h16* wpb = wpad + (size_t)b * 96 * 256;
  const int colLn = ln & 15, rB = (ln >> 4) * 8;

  for (int ct = 0; ct < 16; ++ct) {
    v8f c = {};
#pragma unroll
    for (int kt = 0; kt < 3; ++kt) {
      v16h bm = packB_km(wpb, 256, kt * 32, ct * 16);
      c = wmma16(packA(pT, 96, r0, kt * 32), bm, c);
    }
#pragma unroll
    for (int i = 0; i < 8; ++i) {
      const size_t off = (size_t)(r0 + rB + i) * 256 + ct * 16 + colLn;
      fe[off] = c[i] * fusion[off];
    }
  }
}

// ---------------------------------------------------------------------------
// K8: edge_w[b,j] = sum_l etw[b,l,j]*cw2[b,l]  (deterministic reduction)
// ---------------------------------------------------------------------------
__global__ void edgew_kernel(const float* etwf, float* edge) {
  const int t = threadIdx.x;
  if (t < BQ * 9) {
    const int b = t / 9, j = t - b * 9;
    float s = 0.f;
    for (int l = 0; l < LQ; ++l) s += etwf[(b * LQ + l) * 9 + j];
    edge[t] = s;
  }
}

// ---------------------------------------------------------------------------
extern "C" void kernel_launch(void* const* d_in, const int* in_sizes, int n_in,
                              void* d_out, int out_size, void* d_ws,
                              size_t ws_size, hipStream_t stream) {
  (void)in_sizes; (void)n_in; (void)out_size; (void)ws_size;

  const float* obj   = (const float*)d_in[0];
  const float* lang  = (const float*)d_in[1];
  const float* lmask = (const float*)d_in[2];
  const float* wj1 = (const float*)d_in[3],  *wjb1 = (const float*)d_in[4];
  const float* wj2 = (const float*)d_in[5],  *wjb2 = (const float*)d_in[6];
  const float* eg1 = (const float*)d_in[7],  *egb1 = (const float*)d_in[8];
  const float* eg2 = (const float*)d_in[9],  *egb2 = (const float*)d_in[10];
  const float* lm1 = (const float*)d_in[11], *lmb1 = (const float*)d_in[12];
  const float* lm2 = (const float*)d_in[13], *lmb2 = (const float*)d_in[14];
  const float* wq  = (const float*)d_in[15], *wqb  = (const float*)d_in[16];
  const float* wk  = (const float*)d_in[17], *wkb  = (const float*)d_in[18];
  const float* f1w1 = (const float*)d_in[19], *f1b1 = (const float*)d_in[20];
  const float* f1w2 = (const float*)d_in[21], *f1b2 = (const float*)d_in[22];
  const float* f2w1 = (const float*)d_in[23], *f2b1 = (const float*)d_in[24];
  const float* f2w2 = (const float*)d_in[25], *f2b2 = (const float*)d_in[26];
  const float* f3w1 = (const float*)d_in[27], *f3b1 = (const float*)d_in[28];
  const float* f3w2 = (const float*)d_in[29], *f3b2 = (const float*)d_in[30];

  float* out_fusion = (float*)d_out;
  float* out_edge   = out_fusion + ROWS * 256;
  float* out_fedge  = out_edge + BQ * 9;

  // ---- workspace carve (all sizes multiples of 256 bytes) ----
  char* base = (char*)d_ws;
  size_t off = 0;
  auto take = [&](size_t bytes) -> void* {
    void* p = base + off;
    off += (bytes + 255) & ~(size_t)255;
    return p;
  };
  const size_t pT_bytes = ROWS * 96 * sizeof(h16);          // 12.58 MB each
  h16* pT1  = (h16*)take(pT_bytes);
  h16* pT2  = (h16*)take(pT_bytes);
  h16* pT3  = (h16*)take(pT_bytes);
  h16* wpad = (h16*)take((size_t)BQ * 96 * 256 * sizeof(h16));
  const size_t zero_words = (3 * pT_bytes + (size_t)BQ * 96 * 256 * 2) / 4;
  h16*   v1 = (h16*)take(ROWS * 256 * sizeof(h16));
  h16*   v2 = (h16*)take(ROWS * 256 * sizeof(h16));
  h16*   v3 = (h16*)take(ROWS * 256 * sizeof(h16));
  float* scores = (float*)take((size_t)BQ * LQ * NQ * sizeof(float));
  float* Qp   = (float*)take((size_t)BQ * LQ * 256 * sizeof(float));
  float* qb   = (float*)take((size_t)BQ * LQ * sizeof(float));
  float* wnwm = (float*)take((size_t)BQ * LQ * sizeof(float));
  float* fwm  = (float*)take((size_t)BQ * LQ * sizeof(float));
  float* etwf = (float*)take((size_t)BQ * LQ * 9 * sizeof(float));

  // K0: zero pT pads + words_pad tail
  zero_kernel<<<2048, 256, 0, stream>>>((uint32_t*)pT1, zero_words);

  // K1: language side (cw, etw, words, Q, Qp, qb)
  lang_kernel<<<BQ * LQ, 256, 0, stream>>>(
      lang, lmask, wj1, wjb1, wj2, wjb2, eg1, egb1, eg2, egb2,
      lm1, lmb1, lm2, lmb2, wq, wqb, wk, wkb,
      Qp, qb, wnwm, fwm, etwf, wpad);

  // K3: object MLPs -> v1,v2,v3 (f16)
  vmlp_kernel<<<dim3(512, 3), 256, 0, stream>>>(
      obj, f1w1, f1b1, f1w2, f1b2, f2w1, f2b1, f2w2, f2b2,
      f3w1, f3b1, f3w2, f3b2, v1, v2, v3);

  // attn(v1..v3): scores -> double softmax -> pT_i
  gemm1_kernel<h16><<<512, 256, 0, stream>>>(v1, Qp, qb, scores);
  softmax2_kernel<<<BQ * LQ, 256, 0, stream>>>(scores, wnwm, lmask, pT1);
  gemm1_kernel<h16><<<512, 256, 0, stream>>>(v2, Qp, qb, scores);
  softmax2_kernel<<<BQ * LQ, 256, 0, stream>>>(scores, wnwm, lmask, pT2);
  gemm1_kernel<h16><<<512, 256, 0, stream>>>(v3, Qp, qb, scores);
  softmax2_kernel<<<BQ * LQ, 256, 0, stream>>>(scores, wnwm, lmask, pT3);

  // fusion = sum_i agg_i * v_i
  fuse_kernel<<<512, 256, 0, stream>>>(pT1, pT2, pT3, wpad, v1, v2, v3,
                                       out_fusion);

  // attn(fusion) with cw2 weighting -> fusion_edge
  gemm1_kernel<float><<<512, 256, 0, stream>>>(out_fusion, Qp, qb, scores);
  softmax2_kernel<<<BQ * LQ, 256, 0, stream>>>(scores, fwm, lmask, pT1);
  edgefuse_kernel<<<512, 256, 0, stream>>>(pT1, wpad, out_fusion, out_fedge);

  // edge_w
  edgew_kernel<<<1, 96, 0, stream>>>(etwf, out_edge);
}
